// LiDAR_Renderer_81458349736037
// MI455X (gfx1250) — matrix operand order repaired and checked by reading, not compiled, not run on hardware
//
#include <hip/hip_runtime.h>

// LiDAR volume renderer for MI455X (gfx1250), wave32.
// One wave (32 lanes) per ray; lane = step within a 32-step chunk.
// Transmittance via log-space parallel prefix sum (the +1e-15 is sub-ULP in f32).
// All weighted reductions (20 sem + 2 masked attr + depth + weights_sum) are done
// by two v_wmma_f32_16x16x32_f16 per chunk with a replicated-row A = weights.

typedef __attribute__((ext_vector_type(16))) _Float16     v16h;
typedef __attribute__((ext_vector_type(8)))  float        v8f;
typedef __attribute__((ext_vector_type(4)))  float        v4f;
typedef __attribute__((ext_vector_type(2)))  float        v2f;
typedef __attribute__((ext_vector_type(4)))  unsigned int v4u;

#define NRAYS   8192
#define NSTEPS  768
#define NSEM    20
#define NATTR   2
#define CHUNK   32
#define NCHUNKS (NSTEPS / CHUNK)
// staged channel-row stride in halves: 80 B -> 16B-aligned b128 gathers,
// bank-conflict-free across the 16 reading lanes (20*n mod 64 all distinct)
#define RSTR    40

struct HalfPack { v4u lo; v4u hi; };

__device__ __forceinline__ v16h load_frag(const _Float16* p0, const _Float16* p1) {
  HalfPack hp;
  hp.lo = *(const v4u*)p0;   // 8 halves (16 B, aligned) -> ds_load_b128
  hp.hi = *(const v4u*)p1;   // 8 halves
  return __builtin_bit_cast(v16h, hp);
}

__global__ __launch_bounds__(32)
void lidar_volrender_wmma(const float* __restrict__ sigma,
                          const float* __restrict__ sem,
                          const float* __restrict__ attr,
                          float* __restrict__ out)
{
  constexpr float NEARv = 0.01f, FARv = 0.81f;
  constexpr float DREG  = (FARv - NEARv) / (float)(NSTEPS - 1); // all deltas but last
  constexpr float DLAST = (FARv - NEARv) / (float)NSTEPS;       // appended sample_dist
  constexpr float ZSCL  = (FARv - NEARv) / (float)(NSTEPS - 1); // linspace step

  // B-matrix staging: 32 channel rows x 32 steps (halves), padded stride.
  // rows 0..19: semantic; 20/21: masked attr; 22: z; 23: 1.0; 24..31: zero pad
  __shared__ _Float16 stage[32 * RSTR];
  __shared__ _Float16 wsh[32];           // chunk weights (A-matrix source)

  const int ray  = (int)blockIdx.x;
  const int lane = (int)threadIdx.x;

#pragma unroll
  for (int r = 24; r < 32; ++r) stage[r * RSTR + lane] = (_Float16)0.0f;
  stage[23 * RSTR + lane] = (_Float16)1.0f;   // ones column -> weights_sum
  __syncthreads();

  const float* sigp = sigma + (size_t)ray * NSTEPS;
  const float* semp = sem   + (size_t)ray * NSTEPS * NSEM;
  const float* attp = attr  + (size_t)ray * NSTEPS * NATTR;

  v8f c0 = {};   // columns = sem channels 0..15
  v8f c1 = {};   // columns = sem16..19, attr0, attr1, depth, wsum, 0...
  float carry = 0.0f;   // running sum of delta*sigma across chunks

  // Fragment addressing, contiguous k = 16*(lane/16) + elem for both A and B:
  const int kb = (lane < 16) ? 0 : 16;  // B: K-base for this half-wave
  const int ab = (lane < 16) ? 0 : 16;  // A: K-base (elems 0..15 -> ab..ab+15)
  const int n  = lane & 15;             // B/C column index

  for (int c = 0; c < NCHUNKS; ++c) {
    const int s = c * CHUNK + lane;

    // --- streaming loads (wave reads 128+2560+256 contiguous bytes) ---
    float sg = sigp[s];
    const v4f* sp = (const v4f*)(semp + (size_t)s * NSEM); // 80B/lane, 16B aligned
    v4f q0 = sp[0], q1 = sp[1], q2 = sp[2], q3 = sp[3], q4 = sp[4];
    v2f ap = *(const v2f*)(attp + (size_t)s * NATTR);

    if (c + 1 < NCHUNKS) {  // gfx1250 global_prefetch of next chunk
      __builtin_prefetch(semp + (size_t)(s + CHUNK) * NSEM, 0, 3);
      __builtin_prefetch(sigp + (s + CHUNK), 0, 3);
    }

    // --- transmittance via wave-wide inclusive scan of delta*sigma ---
    float dlt = (s == NSTEPS - 1) ? DLAST : DREG;
    float t = dlt * sg;
    float cum = t;
#pragma unroll
    for (int off = 1; off < 32; off <<= 1) {
      float up = __shfl_up(cum, off);
      cum += (lane >= off) ? up : 0.0f;
    }
    float Texcl = __expf(-(carry + (cum - t)));      // T before this step
    float w = Texcl * (1.0f - __expf(-t));           // alpha * T
    carry += __shfl(cum, 31);                        // chunk total -> next carry

    float z = NEARv + ZSCL * (float)s;
    float m = (w > 1e-4f) ? 1.0f : 0.0f;             // attribute() mask

    // --- stage A (weights) and B (channels) in LDS as f16 ---
    wsh[lane] = (_Float16)w;
#pragma unroll
    for (int k = 0; k < 4; ++k) stage[(0  + k) * RSTR + lane] = (_Float16)q0[k];
#pragma unroll
    for (int k = 0; k < 4; ++k) stage[(4  + k) * RSTR + lane] = (_Float16)q1[k];
#pragma unroll
    for (int k = 0; k < 4; ++k) stage[(8  + k) * RSTR + lane] = (_Float16)q2[k];
#pragma unroll
    for (int k = 0; k < 4; ++k) stage[(12 + k) * RSTR + lane] = (_Float16)q3[k];
#pragma unroll
    for (int k = 0; k < 4; ++k) stage[(16 + k) * RSTR + lane] = (_Float16)q4[k];
    stage[20 * RSTR + lane] = (_Float16)(m * ap[0]); // masked attr0
    stage[21 * RSTR + lane] = (_Float16)(m * ap[1]); // masked attr1
    stage[22 * RSTR + lane] = (_Float16)z;           // z column -> depth

    __syncthreads();  // single-wave WG: hardware NOP, compiler fence

    // A: w replicated across all 16 rows -> every C row holds the partial sums
    v16h a  = load_frag(&wsh[ab],                    &wsh[ab + 8]);
    v16h b0 = load_frag(&stage[n       * RSTR + kb], &stage[n        * RSTR + kb + 8]);
    v16h b1 = load_frag(&stage[(16+n)  * RSTR + kb], &stage[(16 + n) * RSTR + kb + 8]);

    __syncthreads();  // fragments in regs before next chunk overwrites LDS

    c0 = __builtin_amdgcn_wmma_f32_16x16x32_f16(false, a, false, b0, (short)0, c0, false, false);
    c1 = __builtin_amdgcn_wmma_f32_16x16x32_f16(false, a, false, b1, (short)0, c1, false, false);
  }

  // C row 0 lives in VGPR0 of lanes 0..15 (N = lane).
  // out layout: [depth, attr0, attr1, sem0..19, wsum] (24 floats)
  float* o = out + (size_t)ray * 24;
  if (lane < 16) {
    o[3 + lane] = c0[0];                 // sem 0..15
    float v = c1[0];
    if      (lane < 4)  o[19 + lane] = v; // sem 16..19
    else if (lane == 4) o[1]  = v;        // attr0 (masked)
    else if (lane == 5) o[2]  = v;        // attr1 (masked)
    else if (lane == 6) o[0]  = v;        // depth
    else if (lane == 7) o[23] = v;        // weights_sum
  }
}

extern "C" void kernel_launch(void* const* d_in, const int* in_sizes, int n_in,
                              void* d_out, int out_size, void* d_ws, size_t ws_size,
                              hipStream_t stream) {
  (void)in_sizes; (void)n_in; (void)d_ws; (void)ws_size; (void)out_size;
  const float* sigma = (const float*)d_in[0];
  const float* sem   = (const float*)d_in[1];
  const float* attr  = (const float*)d_in[2];
  float* out = (float*)d_out;
  hipLaunchKernelGGL(lidar_volrender_wmma, dim3(NRAYS), dim3(32), 0, stream,
                     sigma, sem, attr, out);
}
